// SelfAttn_17678085390589
// MI455X (gfx1250) — matrix operand order repaired and checked by reading, not compiled, not run on hardware
//
#include <hip/hip_runtime.h>
#include <hip/hip_fp16.h>
#include <stdint.h>

typedef __attribute__((ext_vector_type(16))) _Float16 v16h;
typedef __attribute__((ext_vector_type(4)))  _Float16 v4h;
typedef __attribute__((ext_vector_type(8)))  float    v8f;
typedef __attribute__((ext_vector_type(4)))  float    v4f;

#define HEAD_DIM 128
#define SEQ      2048
#define NBH      32
#define BLK_K    32
#define NBLK     (SEQ / BLK_K)
#define SCALE    0.08838834764831845f   /* 1/sqrt(128) */
#define LN2      0.6931471805599453f
#define LOG2E    1.4426950408889634f

// LDS pitches (halves unless noted)
#define KP 136   // K block row pitch  (32 x 136 halves)
#define VP 40    // V^T row pitch      (128 x 40 halves)
#define PP 40    // P scratch row pitch (per-wave 16 x 40 halves)
#define MP 32    // mask tile row pitch in FLOATS (16B-aligned rows for async b128)

__global__ __launch_bounds__(256, 1)
void SelfAttn_flash_wmma(const float* __restrict__ Q,
                         const float* __restrict__ K,
                         const float* __restrict__ V,
                         const float* __restrict__ Msk,
                         float* __restrict__ Out)
{
    __shared__ _Float16 sK[BLK_K][KP];        // K block, row-major [key][d]
    __shared__ _Float16 sVt[HEAD_DIM][VP];    // V block, transposed [d][key]
    __shared__ _Float16 sP[8][16][PP];        // per-wave P scratch [wave][q][key]
    __shared__ float    sM[8][16][MP];        // per-wave raw mask tile [wave][q][key]

    const int tid  = threadIdx.x;
    const int wave = tid >> 5;
    const int lane = tid & 31;
    const int lh   = lane >> 4;    // half-wave: selects K-range / M-range
    const int ll   = lane & 15;    // N column / A row

    const int bh    = blockIdx.y;
    const int qbase = blockIdx.x * 128 + wave * 16;

    const float* Qb = Q   + ((size_t)bh * SEQ + qbase) * HEAD_DIM;
    const float* Kb = K   + (size_t)bh * SEQ * HEAD_DIM;
    const float* Vb = V   + (size_t)bh * SEQ * HEAD_DIM;
    const float* Mb = Msk + ((size_t)bh * SEQ + qbase) * SEQ;
    float*       Ob = Out + ((size_t)bh * SEQ + qbase) * HEAD_DIM;

    // ---- Q tile -> 4 A-fragments (f32 -> f16), contiguous-k A layout (same k order as B):
    // a[j]   = A[M=ll][c*32 + lh*16 + j]       (j = 0..15)
    v16h qf[4];
    for (int c = 0; c < 4; ++c) {
        const float* src = Qb + (size_t)ll * HEAD_DIM + c * 32 + lh * 16;
        v4f q0 = *(const v4f*)(src);
        v4f q1 = *(const v4f*)(src + 4);
        v4f q2 = *(const v4f*)(src + 8);
        v4f q3 = *(const v4f*)(src + 12);
        for (int j = 0; j < 4; ++j) {
            qf[c][j]      = (_Float16)q0[j];
            qf[c][j + 4]  = (_Float16)q1[j];
            qf[c][j + 8]  = (_Float16)q2[j];
            qf[c][j + 12] = (_Float16)q3[j];
        }
    }

    // ---- accumulators / softmax state (C-layout: row M = r + 8*lh, col N = ll)
    v8f o[8];
    for (int t = 0; t < 8; ++t)
        for (int r = 0; r < 8; ++r) o[t][r] = 0.0f;
    float mrow[8], lrow[8];
    for (int r = 0; r < 8; ++r) { mrow[r] = -3.0e38f; lrow[r] = 0.0f; }

    // ================= software-pipelined prologue =================
    // (a) issue async copy of block 0's mask tile (per-wave LDS buffer)
    for (int i = 0; i < 4; ++i) {
        int f   = i * 32 + lane;
        int mr  = f >> 3;
        int mc4 = f & 7;
        const float* gp = Mb + (size_t)mr * SEQ + mc4 * 4;
        uint32_t lp = (uint32_t)(uintptr_t)&sM[wave][mr][mc4 * 4];
        asm volatile("global_load_async_to_lds_b128 %0, %1, off"
                     :: "v"(lp), "v"((unsigned long long)(uintptr_t)gp)
                     : "memory");
    }
    // (b) prefetch block 0's K/V quads into registers
    v4f kq[4], vq[4];
    for (int i = 0; i < 4; ++i) {
        int f  = i * 256 + tid;
        int kk = f >> 5;
        int d4 = f & 31;
        kq[i] = *(const v4f*)(Kb + (size_t)kk * HEAD_DIM + d4 * 4);
        vq[i] = *(const v4f*)(Vb + (size_t)kk * HEAD_DIM + d4 * 4);
    }

    for (int kb = 0; kb < NBLK; ++kb) {
        const int k0 = kb * BLK_K;

        __syncthreads();   // previous iteration's sK/sVt consumers are done
        // ---- stage current K/V quads (already in regs) as f16 into LDS
        for (int i = 0; i < 4; ++i) {
            int f  = i * 256 + tid;
            int kk = f >> 5;
            int d4 = f & 31;
            v4h kh;
            for (int c = 0; c < 4; ++c) kh[c] = (_Float16)kq[i][c];
            *(v4h*)&sK[kk][d4 * 4] = kh;                       // ds_store_b64
            for (int c = 0; c < 4; ++c)
                sVt[d4 * 4 + c][kk] = (_Float16)vq[i][c];      // transpose
        }
        __syncthreads();

        // ---- prefetch next block's K/V quads (latency hidden by compute)
        if (kb + 1 < NBLK) {
            const int kn = k0 + BLK_K;
            for (int i = 0; i < 4; ++i) {
                int f  = i * 256 + tid;
                int kk = f >> 5;
                int d4 = f & 31;
                kq[i] = *(const v4f*)(Kb + (size_t)(kn + kk) * HEAD_DIM + d4 * 4);
                vq[i] = *(const v4f*)(Vb + (size_t)(kn + kk) * HEAD_DIM + d4 * 4);
            }
        }

        // ---- S = Q * K^T : two 16x16 subtiles, 4 chained K=32 WMMAs each
        v8f st[2];
        for (int nb = 0; nb < 2; ++nb) {
            v8f acc;
            for (int r = 0; r < 8; ++r) acc[r] = 0.0f;
            for (int c = 0; c < 4; ++c) {
                // B layout (32x16): b[j] = B[K = lh*16 + j][N = ll]
                //                        = Ktile[key = ll][d = c*32 + lh*16 + j]
                v16h bf;
                const _Float16* bsrc = &sK[nb * 16 + ll][c * 32 + lh * 16];
                for (int j = 0; j < 16; ++j) bf[j] = bsrc[j];
                acc = __builtin_amdgcn_wmma_f32_16x16x32_f16(
                        false, qf[c], false, bf, (short)0, acc, false, false);
            }
            st[nb] = acc;
        }

        // ---- consume this block's mask tile from LDS
        asm volatile("s_wait_asynccnt 0x0" ::: "memory");
        float mk[16];
        for (int nb = 0; nb < 2; ++nb)
            for (int r = 0; r < 8; ++r)
                mk[nb * 8 + r] = sM[wave][r + lh * 8][nb * 16 + ll];
        // force the sM reads to complete before the async engine may
        // overwrite the buffer (async LDS writes are unordered vs DS loads)
        asm volatile("s_wait_dscnt 0x0" ::: "memory");

        // ---- issue async copy of NEXT block's mask tile
        if (kb + 1 < NBLK) {
            const int kn = k0 + BLK_K;
            for (int i = 0; i < 4; ++i) {
                int f   = i * 32 + lane;
                int mr  = f >> 3;
                int mc4 = f & 7;
                const float* gp = Mb + (size_t)mr * SEQ + (kn + mc4 * 4);
                uint32_t lp = (uint32_t)(uintptr_t)&sM[wave][mr][mc4 * 4];
                asm volatile("global_load_async_to_lds_b128 %0, %1, off"
                             :: "v"(lp), "v"((unsigned long long)(uintptr_t)gp)
                             : "memory");
            }
        }

        // ---- logits = (s + ln(mask)) * scale   (faithful op order)
        for (int nb = 0; nb < 2; ++nb)
            for (int r = 0; r < 8; ++r) {
                float lg = __builtin_amdgcn_logf(mk[nb * 8 + r]) * LN2;
                st[nb][r] = (st[nb][r] + lg) * SCALE;
            }

        // ---- online softmax update (row reductions across 16-lane halves)
        for (int r = 0; r < 8; ++r) {
            float mx = fmaxf(st[0][r], st[1][r]);
            for (int off = 1; off < 16; off <<= 1)
                mx = fmaxf(mx, __shfl_xor(mx, off, 32));
            float mnew  = fmaxf(mrow[r], mx);
            float alpha = __builtin_amdgcn_exp2f((mrow[r] - mnew) * LOG2E);
            float p0 = __builtin_amdgcn_exp2f((st[0][r] - mnew) * LOG2E);
            float p1 = __builtin_amdgcn_exp2f((st[1][r] - mnew) * LOG2E);
            st[0][r] = p0; st[1][r] = p1;
            float rs = p0 + p1;
            for (int off = 1; off < 16; off <<= 1)
                rs += __shfl_xor(rs, off, 32);
            lrow[r] = lrow[r] * alpha + rs;
            mrow[r] = mnew;
            for (int t = 0; t < 8; ++t) o[t][r] *= alpha;
        }

        // ---- P: C-layout regs -> LDS -> A-fragment (f16)
        for (int nb = 0; nb < 2; ++nb)
            for (int r = 0; r < 8; ++r)
                sP[wave][r + lh * 8][nb * 16 + ll] = (_Float16)st[nb][r];
        v16h pf;
        {
            const _Float16* psrc = &sP[wave][ll][lh * 16];
            for (int j = 0; j < 8; ++j) {
                pf[j]     = psrc[j];
                pf[j + 8] = psrc[8 + j];
            }
        }

        // ---- O += P * V : 8 d-tiles, one K=32 WMMA each
        for (int t = 0; t < 8; ++t) {
            // b[j] = B[K = lh*16 + j][N = ll] = V[key = lh*16 + j][d = t*16 + ll]
            v16h vf;
            const _Float16* vsrc = &sVt[t * 16 + ll][lh * 16];
            for (int j = 0; j < 16; ++j) vf[j] = vsrc[j];
            o[t] = __builtin_amdgcn_wmma_f32_16x16x32_f16(
                        false, pf, false, vf, (short)0, o[t], false, false);
        }
    }

    // ---- normalize and store
    for (int t = 0; t < 8; ++t)
        for (int r = 0; r < 8; ++r) {
            float val = o[t][r] / lrow[r];
            Ob[(size_t)(r + lh * 8) * HEAD_DIM + t * 16 + ll] = val;
        }
}

extern "C" void kernel_launch(void* const* d_in, const int* in_sizes, int n_in,
                              void* d_out, int out_size, void* d_ws, size_t ws_size,
                              hipStream_t stream) {
    (void)in_sizes; (void)n_in; (void)out_size; (void)d_ws; (void)ws_size;
    const float* Q   = (const float*)d_in[0];
    const float* K   = (const float*)d_in[1];
    const float* V   = (const float*)d_in[2];
    const float* Msk = (const float*)d_in[3];
    float*       Out = (float*)d_out;

    dim3 grid(SEQ / 128, NBH);   // (16, 32)
    dim3 block(256);             // 8 waves of 32
    hipLaunchKernelGGL(SelfAttn_flash_wmma, grid, block, 0, stream,
                       Q, K, V, Msk, Out);
}
